// NodeContrastiveModel_66683662238188
// MI455X (gfx1250) — matrix operand order repaired and checked by reading, not compiled, not run on hardware
//
#include <hip/hip_runtime.h>
#include <hip/hip_bf16.h>

// N=4096, D=300. K padded to 320 (10 wmma k-steps of 32), N-dim padded to 304.
#define NN 4096
#define DD 300
#define DPN 304
#define KP  320

typedef _Float16 h8   __attribute__((ext_vector_type(8)));
typedef _Float16 v16h __attribute__((ext_vector_type(16)));
typedef float    v8f  __attribute__((ext_vector_type(8)));
typedef float    f4   __attribute__((ext_vector_type(4)));

static __device__ __forceinline__ v16h cat16(h8 lo, h8 hi) {
  return __builtin_shufflevector(lo, hi, 0,1,2,3,4,5,6,7,8,9,10,11,12,13,14,15);
}

// ---------------- utility kernels ----------------
__global__ void zero_kernel(unsigned* p, int n) {
  int i = blockIdx.x * blockDim.x + threadIdx.x;
  if (i < n) p[i] = 0u;
}

__global__ void conv_x_kernel(const float* __restrict__ x, _Float16* __restrict__ x16) {
  int idx = blockIdx.x * blockDim.x + threadIdx.x;
  if (idx < NN * DD) {
    int i = idx / DD, d = idx % DD;
    x16[(size_t)i * KP + d] = (_Float16)x[idx];
  }
}

__global__ void conv_w_kernel(const float* __restrict__ w, _Float16* __restrict__ w16) {
  int idx = blockIdx.x * blockDim.x + threadIdx.x;
  if (idx < DD * DD) {
    int k = idx / DD, n = idx % DD;
    w16[(size_t)k * DPN + n] = (_Float16)w[idx];
  }
}

// ---------------- projector GEMM: C[128x16 per block] = A[NN,KP] * B[KP,DPN] ----------------
// B stored K-row-major (matches W row-major). WMMA B fragment: lane = n, 16 f16 along K.
__global__ __launch_bounds__(256) void proj_gemm(const _Float16* __restrict__ A,
                                                 const _Float16* __restrict__ Bw,
                                                 const float* __restrict__ bias,
                                                 _Float16* __restrict__ Yh,  // mode 0: relu -> f16 [NN][KP]
                                                 float* __restrict__ Yf,    // mode 1: fp32 [NN][DPN]
                                                 int mode) {
  __shared__ __align__(16) _Float16 Bs[KP][16]; // 10 KB
  const int tid = threadIdx.x;
  const int lane = tid & 31, wave = tid >> 5;
  const int hlf = lane >> 4, l15 = lane & 15;
  const int nt = blockIdx.x;          // 0..18
  const int m0 = blockIdx.y * 128;
  for (int idx = tid; idx < KP * 16 / 2; idx += 256) {
    int k = idx >> 3, cc = (idx & 7) << 1;
    *(unsigned*)&Bs[k][cc] = *(const unsigned*)&Bw[(size_t)k * DPN + nt * 16 + cc];
  }
  __syncthreads();
  const int M = m0 + wave * 16 + l15;
  const _Float16* arow = A + (size_t)M * KP;
  v8f acc = (v8f){0.f,0.f,0.f,0.f,0.f,0.f,0.f,0.f};
#pragma unroll
  for (int kk = 0; kk < KP; kk += 32) {
    h8 alo = *(const h8*)(arow + kk + 8 * hlf);
    h8 ahi = *(const h8*)(arow + kk + 16 + 8 * hlf);
    v16h afrag = cat16(alo, ahi);
    v16h bfrag;
#pragma unroll
    for (int s = 0; s < 8; ++s) {
      bfrag[s]     = Bs[kk + 8 * hlf + s][l15];
      bfrag[s + 8] = Bs[kk + 16 + 8 * hlf + s][l15];
    }
    acc = __builtin_amdgcn_wmma_f32_16x16x32_f16(false, afrag, false, bfrag,
                                                 (short)0, acc, false, false);
  }
#pragma unroll
  for (int r = 0; r < 8; ++r) {
    int row = m0 + wave * 16 + r + 8 * hlf;
    int col = nt * 16 + l15;
    if (col < DD) {
      float v = acc[r] + bias[col];
      if (mode == 0) Yh[(size_t)row * KP + col] = (_Float16)fmaxf(v, 0.0f);
      else           Yf[(size_t)row * DPN + col] = v;
    }
  }
}

// ---------------- normalize rows, emit nf [NN][KP] f16 and nfT [KP][NN] f16 ----------------
__global__ void normalize_kernel(const float* __restrict__ Yf,
                                 _Float16* __restrict__ nf, _Float16* __restrict__ nfT) {
  __shared__ float red[128];
  const int i = blockIdx.x, tid = threadIdx.x;
  float s = 0.f;
  for (int d = tid; d < DD; d += 128) { float v = Yf[(size_t)i * DPN + d]; s += v * v; }
  red[tid] = s; __syncthreads();
  for (int off = 64; off; off >>= 1) { if (tid < off) red[tid] += red[tid + off]; __syncthreads(); }
  float inv = 1.0f / fmaxf(sqrtf(red[0]), 1e-12f);
  for (int d = tid; d < DD; d += 128) {
    float v = Yf[(size_t)i * DPN + d] * inv;
    nf [(size_t)i * KP + d] = (_Float16)v;
    nfT[(size_t)d * NN + i] = (_Float16)v;
  }
}

// ---------------- Gram: Kmat = exp(20 * nf nf^T). 128x128 tile/block, 8 waves x (16x128) ----------------
__global__ __launch_bounds__(256) void gram_kernel(const _Float16* __restrict__ nf,
                                                   const _Float16* __restrict__ nfT,
                                                   float* __restrict__ Kmat) {
  __shared__ __align__(16) _Float16 Bs[64][128]; // 16 KB, K-chunk x N-tile
  const int tid = threadIdx.x;
  const int lane = tid & 31, wave = tid >> 5;
  const int hlf = lane >> 4, l15 = lane & 15;
  const int n0 = blockIdx.x * 128, m0 = blockIdx.y * 128;
  const _Float16* arow = nf + (size_t)(m0 + wave * 16 + l15) * KP;
  v8f acc[8];
#pragma unroll
  for (int t = 0; t < 8; ++t) acc[t] = (v8f){0.f,0.f,0.f,0.f,0.f,0.f,0.f,0.f};

  for (int c = 0; c < KP / 64; ++c) {
    __syncthreads();
#pragma unroll
    for (int it = 0; it < 4; ++it) {
      int idx = tid + it * 256;                 // 1024 x 16B chunks
      int k = idx >> 4, c8 = (idx & 15) << 3;
      *(f4*)&Bs[k][c8] = *(const f4*)&nfT[(size_t)(c * 64 + k) * NN + n0 + c8];
    }
    __syncthreads();
#pragma unroll
    for (int k2 = 0; k2 < 2; ++k2) {
      const int kk = c * 64 + k2 * 32;
      h8 alo = *(const h8*)(arow + kk + 8 * hlf);
      h8 ahi = *(const h8*)(arow + kk + 16 + 8 * hlf);
      v16h afrag = cat16(alo, ahi);
      const int kl = k2 * 32 + 8 * hlf;
#pragma unroll
      for (int nt = 0; nt < 8; ++nt) {
        v16h bfrag;
#pragma unroll
        for (int s = 0; s < 8; ++s) {
          bfrag[s]     = Bs[kl + s][nt * 16 + l15];
          bfrag[s + 8] = Bs[kl + 16 + s][nt * 16 + l15];
        }
        acc[nt] = __builtin_amdgcn_wmma_f32_16x16x32_f16(false, afrag, false, bfrag,
                                                         (short)0, acc[nt], false, false);
      }
    }
  }
#pragma unroll
  for (int nt = 0; nt < 8; ++nt)
#pragma unroll
    for (int r = 0; r < 8; ++r) {
      int row = m0 + wave * 16 + r + 8 * hlf;
      int col = n0 + nt * 16 + l15;
      Kmat[(size_t)row * NN + col] = __expf(20.0f * acc[nt][r]); // 1/T = 20
    }
}

// ---------------- row sums: lse_i = log(sum K), u_i = 1/sum  (first Sinkhorn half-step) ----------------
__global__ void rowsum_kernel(const float* __restrict__ Kmat, float* __restrict__ u,
                              float* __restrict__ lse) {
  __shared__ float red[256];
  const int i = blockIdx.x, tid = threadIdx.x;
  float s = 0.f;
  for (int j = tid; j < NN; j += 256) s += Kmat[(size_t)i * NN + j];
  red[tid] = s; __syncthreads();
  for (int off = 128; off; off >>= 1) { if (tid < off) red[tid] += red[tid + off]; __syncthreads(); }
  if (tid == 0) { u[i] = 1.0f / red[0]; lse[i] = __logf(red[0]); }
}

__global__ void kv_kernel(const float* __restrict__ Kmat, const float* __restrict__ v,
                          float* __restrict__ u) {
  __shared__ float red[256];
  const int i = blockIdx.x, tid = threadIdx.x;
  float s = 0.f;
  for (int j = tid; j < NN; j += 256) s += Kmat[(size_t)i * NN + j] * v[j];
  red[tid] = s; __syncthreads();
  for (int off = 128; off; off >>= 1) { if (tid < off) red[tid] += red[tid + off]; __syncthreads(); }
  if (tid == 0) u[i] = 1.0f / red[0];
}

__global__ void uk_kernel(const float* __restrict__ Kmat, const float* __restrict__ u,
                          float* __restrict__ v) {
  int j = blockIdx.x * blockDim.x + threadIdx.x;   // coalesced across columns
  float s = 0.f;
  for (int i = 0; i < NN; ++i) s += u[i] * Kmat[(size_t)i * NN + j];
  v[j] = 1.0f / s;
}

// ---------------- sparse 2-hop adjacency via bitmaps (replaces dense 137-GFLOP adj@adj) ----------------
__global__ void edges_kernel(const int* __restrict__ ei, int E, unsigned* __restrict__ adjb) {
  int idx = blockIdx.x * blockDim.x + threadIdx.x;
  if (idx < E) {
    int s = ei[idx], d = ei[E + idx];
    atomicOr(&adjb[(size_t)s * 128 + (d >> 5)], 1u << (d & 31));
  } else if (idx < E + NN) {
    int i = idx - E;
    atomicOr(&adjb[(size_t)i * 128 + (i >> 5)], 1u << (i & 31));
  }
}

__global__ void adj2_kernel(const unsigned* __restrict__ adjb, unsigned* __restrict__ adj2b) {
  __shared__ unsigned rowi[128];
  const int i = blockIdx.x, tid = threadIdx.x;
  rowi[tid] = adjb[(size_t)i * 128 + tid];
  __syncthreads();
  unsigned acc = 0;
  for (int w = 0; w < 128; ++w) {
    unsigned b = rowi[w];
    while (b) {
      int k = w * 32 + __builtin_ctz(b);
      b &= b - 1;
      acc |= adjb[(size_t)k * 128 + tid];
    }
  }
  adj2b[(size_t)i * 128 + tid] = acc;
}

__global__ void colsum_kernel(const unsigned* __restrict__ adj2b, float* __restrict__ colsum) {
  int j = blockIdx.x * blockDim.x + threadIdx.x;
  int w = j >> 5, b = j & 31;
  int cnt = 0;
  for (int i = 0; i < NN; ++i) cnt += (adj2b[(size_t)i * 128 + w] >> b) & 1;
  colsum[j] = (float)cnt;
}

// ---------------- fused loss: sum_j (u_i K_ij v_j + bit_ij/colsum_j) * (lse_i - log K_ij) ----------------
__global__ void loss_kernel(const float* __restrict__ Kmat, const float* __restrict__ u,
                            const float* __restrict__ v, const float* __restrict__ lse,
                            const unsigned* __restrict__ adj2b, const float* __restrict__ colsum,
                            float* __restrict__ rowpart) {
  __shared__ float red[256];
  const int i = blockIdx.x, tid = threadIdx.x;
  const float ui = u[i], li = lse[i];
  float s = 0.f;
  for (int j = tid; j < NN; j += 256) {
    float Kij = Kmat[(size_t)i * NN + j];
    float nlp = li - __logf(Kij);                        // -(log_probs)
    float coup = ui * Kij * v[j];
    unsigned bit = (adj2b[(size_t)i * 128 + (j >> 5)] >> (j & 31)) & 1u;
    float tgt = bit ? (1.0f / colsum[j]) : 0.0f;
    s += (coup + tgt) * nlp;
  }
  red[tid] = s; __syncthreads();
  for (int off = 128; off; off >>= 1) { if (tid < off) red[tid] += red[tid + off]; __syncthreads(); }
  if (tid == 0) rowpart[i] = red[0];
}

__global__ void final_reduce(const float* __restrict__ rowpart, float* __restrict__ out) {
  __shared__ float red[256];
  const int tid = threadIdx.x;
  float s = 0.f;
  for (int i = tid; i < NN; i += 256) s += rowpart[i];
  red[tid] = s; __syncthreads();
  for (int off = 128; off; off >>= 1) { if (tid < off) red[tid] += red[tid + off]; __syncthreads(); }
  if (tid == 0) out[0] = red[0] * (1.0f / (float)NN);
}

// ---------------- host ----------------
extern "C" void kernel_launch(void* const* d_in, const int* in_sizes, int n_in,
                              void* d_out, int out_size, void* d_ws, size_t ws_size,
                              hipStream_t stream) {
  (void)n_in; (void)out_size; (void)ws_size;
  const float* x  = (const float*)d_in[0];
  const float* W1 = (const float*)d_in[1];
  const float* b1 = (const float*)d_in[2];
  const float* W2 = (const float*)d_in[3];
  const float* b2 = (const float*)d_in[4];
  const int*   ei = (const int*)d_in[5];
  const int E = in_sizes[5] / 2;
  float* out = (float*)d_out;

  char* p = (char*)d_ws;
  auto alloc = [&](size_t bytes) -> void* {
    void* r = (void*)p; p += (bytes + 255) & ~(size_t)255; return r;
  };
  float*    Kmat  = (float*)alloc((size_t)NN * NN * 4);      // 67 MB, lives in L2
  _Float16* x16   = (_Float16*)alloc((size_t)NN * KP * 2);
  _Float16* y16   = (_Float16*)alloc((size_t)NN * KP * 2);
  _Float16* w116  = (_Float16*)alloc((size_t)KP * DPN * 2);
  _Float16* w216  = (_Float16*)alloc((size_t)KP * DPN * 2);
  float*    out32 = (float*)alloc((size_t)NN * DPN * 4);
  _Float16* nf    = (_Float16*)alloc((size_t)NN * KP * 2);
  _Float16* nfT   = (_Float16*)alloc((size_t)KP * NN * 2);
  float*    u     = (float*)alloc(NN * 4);
  float*    v     = (float*)alloc(NN * 4);
  float*    lse   = (float*)alloc(NN * 4);
  float*    rowp  = (float*)alloc(NN * 4);
  float*    csum  = (float*)alloc(NN * 4);
  unsigned* adjb  = (unsigned*)alloc((size_t)NN * 128 * 4);
  unsigned* adj2b = (unsigned*)alloc((size_t)NN * 128 * 4);

  auto zero = [&](void* q, size_t bytes) {
    int n = (int)(bytes / 4);
    zero_kernel<<<(n + 255) / 256, 256, 0, stream>>>((unsigned*)q, n);
  };
  zero(x16,  (size_t)NN * KP * 2);
  zero(y16,  (size_t)NN * KP * 2);
  zero(w116, (size_t)KP * DPN * 2);
  zero(w216, (size_t)KP * DPN * 2);
  zero(nf,   (size_t)NN * KP * 2);
  zero(nfT,  (size_t)KP * NN * 2);
  zero(adjb, (size_t)NN * 128 * 4);

  conv_x_kernel<<<(NN * DD + 255) / 256, 256, 0, stream>>>(x, x16);
  conv_w_kernel<<<(DD * DD + 255) / 256, 256, 0, stream>>>(W1, w116);
  conv_w_kernel<<<(DD * DD + 255) / 256, 256, 0, stream>>>(W2, w216);

  proj_gemm<<<dim3(DPN / 16, NN / 128), 256, 0, stream>>>(x16, w116, b1, y16, nullptr, 0);
  proj_gemm<<<dim3(DPN / 16, NN / 128), 256, 0, stream>>>(y16, w216, b2, nullptr, out32, 1);
  normalize_kernel<<<NN, 128, 0, stream>>>(out32, nf, nfT);

  gram_kernel<<<dim3(NN / 128, NN / 128), 256, 0, stream>>>(nf, nfT, Kmat);

  rowsum_kernel<<<NN, 256, 0, stream>>>(Kmat, u, lse); // lse + Sinkhorn iter-1 u (v==1)
  for (int t = 0; t < 10; ++t) {
    if (t) kv_kernel<<<NN, 256, 0, stream>>>(Kmat, v, u);
    uk_kernel<<<NN / 256, 256, 0, stream>>>(Kmat, u, v);
  }

  edges_kernel<<<(E + NN + 255) / 256, 256, 0, stream>>>(ei, E, adjb);
  adj2_kernel<<<NN, 128, 0, stream>>>(adjb, adj2b);
  colsum_kernel<<<NN / 256, 256, 0, stream>>>(adj2b, csum);

  loss_kernel<<<NN, 256, 0, stream>>>(Kmat, u, v, lse, adj2b, csum, rowp);
  final_reduce<<<1, 256, 0, stream>>>(rowp, out);
}
